// SS2D_68513318305916
// MI455X (gfx1250) — matrix operand order repaired and checked by reading, not compiled, not run on hardware
//
#include <hip/hip_runtime.h>
#include <hip/hip_bf16.h>
#include <stdint.h>

typedef __attribute__((ext_vector_type(16))) _Float16 v16h;
typedef __attribute__((ext_vector_type(8)))  _Float16 v8h;
typedef __attribute__((ext_vector_type(8)))  float    v8f;

#define B_    4
#define H_    64
#define W_    64
#define L_    4096
#define BL_   16384      // B_*L_
#define DM_   192
#define DI_   384
#define NS_   16         // D_STATE
#define DTR_  12
#define K_    4
#define C44   44         // DT_RANK + 2*D_STATE
#define C48   48         // padded
#define SCAN_D 128

// ---------------------------------------------------------------------------
// fp32 -> fp16 convert
// ---------------------------------------------------------------------------
__global__ void k_f32_to_f16(const float* __restrict__ src,
                             _Float16* __restrict__ dst, int n) {
  int i = blockIdx.x * 256 + threadIdx.x;
  if (i < n) dst[i] = (_Float16)src[i];
}

// x_proj_w (K,44,384) -> padded f16 (K,48,384), pad rows zero
__global__ void k_conv_xprojw(const float* __restrict__ src,
                              _Float16* __restrict__ dst) {
  int i = blockIdx.x * 256 + threadIdx.x;
  if (i >= K_ * C48 * DI_) return;
  int d = i % DI_;
  int c = (i / DI_) % C48;
  int k = i / (DI_ * C48);
  float v = (c < C44) ? src[((size_t)k * C44 + c) * DI_ + d] : 0.0f;
  dst[i] = (_Float16)v;
}

// ---------------------------------------------------------------------------
// WMMA 16x64 tile: D(16x64,f32) += A(16xK,f16 row-major) * B(Kx64,f16 row-major)
// A frag: lane = M row; halves 0..7 = K(kh*8 ..), halves 8..15 = K(16+kh*8 ..)
// B frag: lane = N column; halves 0..7 = K(kh*8 ..), halves 8..15 = K(16+kh*8 ..)
// ---------------------------------------------------------------------------
__device__ __forceinline__ void wmma_tile_16x64(
    const _Float16* __restrict__ A, int lda,
    const _Float16* __restrict__ Bm, int ldb,
    int m0, int n0, int K, v8f acc[4]) {
  const int lane = threadIdx.x & 31;
  const int m    = lane & 15;
  const int kh   = (lane >> 4) << 3;             // 0 or 8
  const _Float16* arow = A + (size_t)(m0 + m) * lda;
  for (int k0 = 0; k0 < K; k0 += 32) {
    v8h lo = *(const v8h*)(arow + k0 + kh);
    v8h hi = *(const v8h*)(arow + k0 + 16 + kh);
    v16h a;
#pragma unroll
    for (int i = 0; i < 8; ++i) { a[i] = lo[i]; a[8 + i] = hi[i]; }
    const _Float16* bp = Bm + (size_t)(k0 + kh) * ldb + n0 + m;
#pragma unroll
    for (int j = 0; j < 4; ++j) {
      v16h bfr;
#pragma unroll
      for (int i = 0; i < 8; ++i) {
        bfr[i] = bp[(size_t)i * ldb + j * 16];
        bfr[8 + i] = bp[(size_t)(16 + i) * ldb + j * 16];
      }
      acc[j] = __builtin_amdgcn_wmma_f32_16x16x32_f16(
          false, a, false, bfr, (short)0, acc[j], false, false);
    }
  }
}

// ---------------------------------------------------------------------------
// in_proj: X(16384x192) @ W(192x768) -> xb(:, :384), z(:, 384:)
// ---------------------------------------------------------------------------
__global__ void __launch_bounds__(256) k_gemm_inproj(
    const _Float16* __restrict__ A, const _Float16* __restrict__ Bm,
    float* __restrict__ xb, float* __restrict__ zg) {
  int wid = blockIdx.x * 8 + (threadIdx.x >> 5);      // 12288 tiles
  int mt = wid / 12, nt = wid % 12;
  int m0 = mt * 16, n0 = nt * 64;
  v8f acc[4] = {};
  wmma_tile_16x64(A, DM_, Bm, 2 * DI_, m0, n0, DM_, acc);
  int lane = threadIdx.x & 31;
  int mrow = m0 + ((lane >> 4) << 3);
  float* outp = (n0 < DI_) ? xb : zg;
  int nb = (n0 < DI_) ? n0 : (n0 - DI_);
#pragma unroll
  for (int j = 0; j < 4; ++j) {
    int n = nb + j * 16 + (lane & 15);
#pragma unroll
    for (int r = 0; r < 8; ++r)
      outp[(size_t)(mrow + r) * DI_ + n] = acc[j][r];
  }
}

// ---------------------------------------------------------------------------
// depthwise 3x3 conv (SAME) + SiLU; layout (b, hw, d)
// ---------------------------------------------------------------------------
__global__ void k_dwconv_silu(const float* __restrict__ xb,
                              const float* __restrict__ cw,
                              const float* __restrict__ cb,
                              float* __restrict__ xc) {
  size_t i = (size_t)blockIdx.x * 256 + threadIdx.x;
  if (i >= (size_t)BL_ * DI_) return;
  int d  = (int)(i % DI_);
  int hw = (int)((i / DI_) % L_);
  int b  = (int)(i / ((size_t)DI_ * L_));
  int h = hw >> 6, w = hw & 63;
  float acc = cb[d];
#pragma unroll
  for (int kh = 0; kh < 3; ++kh) {
    int nh = h + kh - 1;
    if (nh < 0 || nh >= H_) continue;
#pragma unroll
    for (int kw = 0; kw < 3; ++kw) {
      int nw = w + kw - 1;
      if (nw < 0 || nw >= W_) continue;
      acc += cw[d * 9 + kh * 3 + kw] *
             xb[((size_t)b * L_ + nh * W_ + nw) * DI_ + d];
    }
  }
  xc[i] = acc / (1.0f + __expf(-acc));   // SiLU
}

// direction k, scan index l -> canonical hw index
__device__ __forceinline__ int scan_map(int k, int l) {
  switch (k) {
    case 0:  return l;
    case 1:  return ((l & 63) << 6) | (l >> 6);         // transpose
    case 2:  return L_ - 1 - l;                          // flip
    default: { int l2 = L_ - 1 - l; return ((l2 & 63) << 6) | (l2 >> 6); }
  }
}

// build xs f16, layout (k, d, b*L + l) scan-ordered (GEMM B operand)
__global__ void k_cross_scan(const float* __restrict__ xc,
                             _Float16* __restrict__ xs) {
  size_t i = (size_t)blockIdx.x * 256 + threadIdx.x;   // K*DI*B*L
  if (i >= (size_t)K_ * DI_ * B_ * L_) return;
  int l = (int)(i % L_);
  size_t t = i / L_;
  int b = (int)(t % B_); t /= B_;
  int d = (int)(t % DI_);
  int k = (int)(t / DI_);
  int lc = scan_map(k, l);
  xs[i] = (_Float16)xc[((size_t)b * L_ + lc) * DI_ + d];
}

// ---------------------------------------------------------------------------
// x_proj: per k: W(48x384) @ xs_k(384x16384) -> xdbl (k, bl, 48)
// ---------------------------------------------------------------------------
__global__ void __launch_bounds__(256) k_gemm_xproj(
    const _Float16* __restrict__ Wp, const _Float16* __restrict__ xs,
    float* __restrict__ xdbl) {
  int wid = blockIdx.x * 8 + (threadIdx.x >> 5);  // 3072 = K_*3*256
  int nt = wid % 256;
  int t  = wid / 256;
  int mt = t % 3;
  int kk = t / 3;
  int m0 = mt * 16, n0 = nt * 64;
  const _Float16* A  = Wp + (size_t)kk * C48 * DI_;
  const _Float16* Bm = xs + (size_t)kk * DI_ * BL_;
  v8f acc[4] = {};
  wmma_tile_16x64(A, DI_, Bm, BL_, m0, n0, DI_, acc);
  int lane = threadIdx.x & 31;
  int c = m0 + ((lane >> 4) << 3);
#pragma unroll
  for (int j = 0; j < 4; ++j) {
    int bl = n0 + j * 16 + (lane & 15);
    float* dst = xdbl + ((size_t)kk * BL_ + bl) * C48 + c;
#pragma unroll
    for (int r = 0; r < 8; ++r) dst[r] = acc[j][r];
  }
}

// ---------------------------------------------------------------------------
// dt_proj + softplus -> delta (k, b, l, d)
// ---------------------------------------------------------------------------
__global__ void __launch_bounds__(DI_) k_dtproj(
    const float* __restrict__ xdbl, const float* __restrict__ dtw,
    const float* __restrict__ dtb, float* __restrict__ delta) {
  int blk = blockIdx.x;                 // K_*B_*L_
  int l = blk % L_;
  int b = (blk / L_) % B_;
  int k = blk / (L_ * B_);
  int d = threadIdx.x;
  __shared__ float r[DTR_];
  const float* src = xdbl + ((size_t)k * BL_ + (size_t)b * L_ + l) * C48;
  if (d < DTR_) r[d] = src[d];
  __syncthreads();
  float acc = dtb[k * DI_ + d];
  const float* w = dtw + ((size_t)k * DI_ + d) * DTR_;
#pragma unroll
  for (int j = 0; j < DTR_; ++j) acc += w[j] * r[j];
  float sp = (acc > 20.0f) ? acc : log1pf(__expf(acc));
  delta[(((size_t)k * B_ + b) * L_ + l) * DI_ + d] = sp;
}

// ---------------------------------------------------------------------------
// selective scan: 48 blocks = (b,k,d-chunk of 128); h[16] in registers.
// Writes y at canonical positions so directions merge additively later.
// ---------------------------------------------------------------------------
__global__ void __launch_bounds__(SCAN_D) k_selective_scan(
    const float* __restrict__ xc, const float* __restrict__ xdbl,
    const float* __restrict__ delta, const float* __restrict__ A_logs,
    float* __restrict__ oy) {
  const int nchunk = DI_ / SCAN_D;
  int chunk = blockIdx.x % nchunk;
  int k = (blockIdx.x / nchunk) % K_;
  int b = blockIdx.x / (nchunk * K_);
  int d = chunk * SCAN_D + threadIdx.x;

  float a[NS_], h[NS_];
#pragma unroll
  for (int n = 0; n < NS_; ++n) {
    a[n] = -__expf(A_logs[((size_t)k * DI_ + d) * NS_ + n]);
    h[n] = 0.0f;
  }
  __shared__ float sBC[2 * NS_];
  const float* dlt = delta + (((size_t)k * B_ + b) * L_) * DI_ + d;
  const float* xdb = xdbl + ((size_t)k * BL_ + (size_t)b * L_) * C48;
  const float* xcb = xc + (size_t)b * L_ * DI_ + d;
  float* oyb = oy + (((size_t)k * B_ + b) * L_) * DI_ + d;

  for (int l = 0; l < L_; ++l) {
    int lc = scan_map(k, l);
    if (threadIdx.x < 2 * NS_)
      sBC[threadIdx.x] = xdb[(size_t)l * C48 + DTR_ + threadIdx.x];
    __syncthreads();
    float de = dlt[(size_t)l * DI_];
    float u  = xcb[(size_t)lc * DI_];
    float du = de * u;
    float y = 0.0f;
#pragma unroll
    for (int n = 0; n < NS_; ++n) {
      h[n] = h[n] * __expf(de * a[n]) + du * sBC[n];
      y += h[n] * sBC[NS_ + n];
    }
    oyb[(size_t)lc * DI_] = y;
    __syncthreads();
  }
}

// ---------------------------------------------------------------------------
// merge 4 directions + D*u + LayerNorm + SiLU gate -> f16 (bl, d)
// ---------------------------------------------------------------------------
__global__ void __launch_bounds__(DI_) k_combine_ln_gate(
    const float* __restrict__ oy, const float* __restrict__ xc,
    const float* __restrict__ Ds, const float* __restrict__ lnw,
    const float* __restrict__ lnb, const float* __restrict__ zg,
    _Float16* __restrict__ yg) {
  int bl = blockIdx.x;                  // BL_
  int b = bl / L_, l = bl % L_;
  int d = threadIdx.x;
  float sumDs = Ds[d] + Ds[DI_ + d] + Ds[2 * DI_ + d] + Ds[3 * DI_ + d];
  float v = xc[(size_t)bl * DI_ + d] * sumDs;
#pragma unroll
  for (int k = 0; k < K_; ++k)
    v += oy[(((size_t)k * B_ + b) * L_ + l) * DI_ + d];

  __shared__ float s1[DI_], s2[DI_];
  s1[d] = v; s2[d] = v * v;
  __syncthreads();
  for (int s = 256; s > 0; s >>= 1) {
    if (d < s && d + s < DI_) { s1[d] += s1[d + s]; s2[d] += s2[d + s]; }
    __syncthreads();
  }
  float mu  = s1[0] * (1.0f / DI_);
  float var = s2[0] * (1.0f / DI_) - mu * mu;
  float nv = (v - mu) * rsqrtf(var + 1e-5f) * lnw[d] + lnb[d];
  float zz = zg[(size_t)bl * DI_ + d];
  float g  = zz / (1.0f + __expf(-zz));
  yg[(size_t)bl * DI_ + d] = (_Float16)(nv * g);
}

// ---------------------------------------------------------------------------
// out_proj: Y(16384x384) @ Wout(384x192) -> out f32
// ---------------------------------------------------------------------------
__global__ void __launch_bounds__(256) k_gemm_outproj(
    const _Float16* __restrict__ A, const _Float16* __restrict__ Bm,
    float* __restrict__ out) {
  int wid = blockIdx.x * 8 + (threadIdx.x >> 5);  // 3072 = 1024*3
  int mt = wid / 3, nt = wid % 3;
  int m0 = mt * 16, n0 = nt * 64;
  v8f acc[4] = {};
  wmma_tile_16x64(A, DI_, Bm, DM_, m0, n0, DI_, acc);
  int lane = threadIdx.x & 31;
  int mrow = m0 + ((lane >> 4) << 3);
#pragma unroll
  for (int j = 0; j < 4; ++j) {
    int n = n0 + j * 16 + (lane & 15);
#pragma unroll
    for (int r = 0; r < 8; ++r)
      out[(size_t)(mrow + r) * DM_ + n] = acc[j][r];
  }
}

// ---------------------------------------------------------------------------
extern "C" void kernel_launch(void* const* d_in, const int* in_sizes, int n_in,
                              void* d_out, int out_size, void* d_ws,
                              size_t ws_size, hipStream_t stream) {
  (void)in_sizes; (void)n_in; (void)out_size; (void)ws_size;
  const float* x      = (const float*)d_in[0];
  const float* w_in   = (const float*)d_in[1];
  const float* conv_w = (const float*)d_in[2];
  const float* conv_b = (const float*)d_in[3];
  const float* xprojw = (const float*)d_in[4];
  const float* dtw    = (const float*)d_in[5];
  const float* dtb    = (const float*)d_in[6];
  const float* A_logs = (const float*)d_in[7];
  const float* Ds     = (const float*)d_in[8];
  const float* lnw    = (const float*)d_in[9];
  const float* lnb    = (const float*)d_in[10];
  const float* outw   = (const float*)d_in[11];
  float* out = (float*)d_out;

  char* ws = (char*)d_ws;
  size_t off = 0;
  auto carve = [&](size_t bytes) -> void* {
    void* p = ws + off;
    off = (off + bytes + 255) & ~(size_t)255;
    return p;
  };
  _Float16* Xf16  = (_Float16*)carve((size_t)BL_ * DM_ * 2);
  _Float16* W1f16 = (_Float16*)carve((size_t)DM_ * 2 * DI_ * 2);
  _Float16* XPW16 = (_Float16*)carve((size_t)K_ * C48 * DI_ * 2);
  _Float16* WO16  = (_Float16*)carve((size_t)DI_ * DM_ * 2);
  float* xb    = (float*)carve((size_t)BL_ * DI_ * 4);
  float* zgate = (float*)carve((size_t)BL_ * DI_ * 4);
  float* xc    = (float*)carve((size_t)BL_ * DI_ * 4);
  _Float16* xs16 = (_Float16*)carve((size_t)K_ * DI_ * BL_ * 2);
  float* xdbl  = (float*)carve((size_t)K_ * BL_ * C48 * 4);
  float* delta = (float*)carve((size_t)K_ * BL_ * DI_ * 4);
  float* oy    = (float*)carve((size_t)K_ * BL_ * DI_ * 4);
  _Float16* yg16 = (_Float16*)carve((size_t)BL_ * DI_ * 2);

  int n;
  n = BL_ * DM_;
  k_f32_to_f16<<<(n + 255) / 256, 256, 0, stream>>>(x, Xf16, n);
  n = DM_ * 2 * DI_;
  k_f32_to_f16<<<(n + 255) / 256, 256, 0, stream>>>(w_in, W1f16, n);
  k_conv_xprojw<<<(K_ * C48 * DI_ + 255) / 256, 256, 0, stream>>>(xprojw, XPW16);
  n = DI_ * DM_;
  k_f32_to_f16<<<(n + 255) / 256, 256, 0, stream>>>(outw, WO16, n);

  k_gemm_inproj<<<1536, 256, 0, stream>>>(Xf16, W1f16, xb, zgate);

  {
    size_t tot = (size_t)BL_ * DI_;
    k_dwconv_silu<<<(unsigned)((tot + 255) / 256), 256, 0, stream>>>(
        xb, conv_w, conv_b, xc);
  }
  {
    size_t tot = (size_t)K_ * DI_ * B_ * L_;
    k_cross_scan<<<(unsigned)((tot + 255) / 256), 256, 0, stream>>>(xc, xs16);
  }

  k_gemm_xproj<<<384, 256, 0, stream>>>(XPW16, xs16, xdbl);
  k_dtproj<<<K_ * B_ * L_, DI_, 0, stream>>>(xdbl, dtw, dtb, delta);
  k_selective_scan<<<B_ * K_ * (DI_ / SCAN_D), SCAN_D, 0, stream>>>(
      xc, xdbl, delta, A_logs, oy);
  k_combine_ln_gate<<<BL_, DI_, 0, stream>>>(oy, xc, Ds, lnw, lnb, zgate, yg16);
  k_gemm_outproj<<<384, 256, 0, stream>>>(yg16, WO16, out);
}
